// LatentODE_84421877170853
// MI455X (gfx1250) — matrix operand order repaired and checked by reading, not compiled, not run on hardware
//
#include <hip/hip_runtime.h>
#include <hip/hip_bf16.h>
#include <math.h>

#define B_  64
#define T_  1000
#define N_  128
#define H_  128
#define L_  64
#define WD_ 256
#define C_  8
#define BH_ 4

typedef __attribute__((ext_vector_type(16))) __bf16 v16bf;
typedef __attribute__((ext_vector_type(8)))  float  v8f;

// ---- helpers -------------------------------------------------------------

__device__ __forceinline__ __bf16 f2bf(float x) {
  union { float f; unsigned u; } v; v.f = x;
  unsigned r = v.u + 0x7FFFu + ((v.u >> 16) & 1u);   // round-to-nearest-even
  unsigned short h = (unsigned short)(r >> 16);
  __bf16 o; __builtin_memcpy(&o, &h, 2); return o;
}

__device__ __forceinline__ v8f vzero() {
  v8f z;
#pragma unroll
  for (int i = 0; i < 8; ++i) z[i] = 0.f;
  return z;
}

// A-matrix 16x32 bf16 swizzle (ISA 7.12.2): lane = m + 16*((k>>3)&1),
// elem = (k&7) + 8*(k>>4). Stored lane-major so a fragment is one
// contiguous 32B read per lane.
__device__ __forceinline__ int a_sw(int tile, int m, int k) {
  int lane = m + (((k >> 3) & 1) << 4);
  int e    = (k & 7) + ((k >> 4) << 3);
  return tile * 512 + (lane << 4) + e;
}
// B-matrix 32x16 bf16 swizzle: lane = n + 16*((k>>3)&1), elem = (k&7) + 8*(k>>4).
__device__ __forceinline__ int b_sw(int tile, int k, int n) {
  int lane = n + (((k >> 3) & 1) << 4);
  return tile * 512 + (lane << 4) + (k & 7) + ((k >> 4) << 3);
}
__device__ __forceinline__ v16bf ldfrag(const __bf16* base, int tile) {
  return *(const v16bf*)(base + tile * 512 + (threadIdx.x & 31) * 16);
}
__device__ __forceinline__ v8f wmma_bf16(v16bf a, v16bf b, v8f c) {
  return __builtin_amdgcn_wmma_f32_16x16x32_bf16(false, a, false, b,
                                                 (short)0, c, false, false);
}

// ---- kernel 1: GRU encoder over reversed sequence ------------------------
// 4 workgroups x 16 batch rows. Per step: two [16,128]x[128,384] bf16 WMMA
// GEMMs (Wx_y and Wh), t-column handled as scalar rank-1 update.

__global__ void __launch_bounds__(256) gru_encoder_kernel(
    const float* __restrict__ ts, const float* __restrict__ ys,
    const float* __restrict__ Wx, const float* __restrict__ Wh,
    const float* __restrict__ eb, const float* __restrict__ h2lW,
    const float* __restrict__ h2lb, float* __restrict__ y0out)
{
  extern __shared__ __align__(32) char smem_raw[];
  char* p = smem_raw;
  auto alloc = [&](size_t n) { char* r = p; p += (n + 31) & ~size_t(31); return r; };
  __bf16* sWx  = (__bf16*)alloc(4 * 24 * 512 * 2);  // B-swizzle K=128,N=384
  __bf16* sWh  = (__bf16*)alloc(4 * 24 * 512 * 2);
  float*  sGx  = (float*)alloc(16 * 384 * 4);
  float*  sGh  = (float*)alloc(16 * 384 * 4);
  __bf16* sAy  = (__bf16*)alloc(4 * 512 * 2);       // A-swizzle y_t [16,128]
  __bf16* sAh  = (__bf16*)alloc(4 * 512 * 2);       // A-swizzle h   [16,128]
  float*  shst = (float*)alloc(16 * 128 * 4);       // fp32 hidden state
  float*  sb   = (float*)alloc(384 * 4);
  float*  swt  = (float*)alloc(384 * 4);            // Wx[:,0] (t column)

  const int tid = threadIdx.x;
  const int wave = tid >> 5, lane = tid & 31;
  const int b0 = blockIdx.x * 16;

  for (int i = tid; i < 384 * 128; i += 256) {
    int n = i >> 7, k = i & 127;
    int t = (k >> 5) * 24 + (n >> 4);
    sWx[b_sw(t, k & 31, n & 15)] = f2bf(Wx[n * 129 + 1 + k]);
    sWh[b_sw(t, k & 31, n & 15)] = f2bf(Wh[n * 128 + k]);
  }
  for (int i = tid; i < 384; i += 256) { sb[i] = eb[i]; swt[i] = Wx[i * 129]; }
  for (int i = tid; i < 16 * 128; i += 256) shst[i] = 0.f;
  for (int i = tid; i < 4 * 512; i += 256) sAh[i] = f2bf(0.f);  // h0 = 0
  __syncthreads();

  for (int step = 0; step < T_; ++step) {
    int rev = T_ - 1 - step;  // time-reversed input
    for (int i = tid; i < 16 * 128; i += 256) {
      int m = i >> 7, k = i & 127;
      size_t base = (size_t)(b0 + m) * ((size_t)T_ * N_) + (size_t)rev * N_ + k;
      sAy[a_sw(k >> 5, m, k & 31)] = f2bf(ys[base]);
      if (rev > 0) __builtin_prefetch(&ys[base - N_], 0, 1);
    }
    __syncthreads();

    // GEMMs: 24 n-tiles over 8 waves (3 each), K = 128 (4 k-steps)
    v8f accx[3], acch[3];
#pragma unroll
    for (int q = 0; q < 3; ++q) { accx[q] = vzero(); acch[q] = vzero(); }
#pragma unroll
    for (int kt = 0; kt < 4; ++kt) {
      v16bf fy = ldfrag(sAy, kt);
      v16bf fh = ldfrag(sAh, kt);
#pragma unroll
      for (int q = 0; q < 3; ++q) {
        int nt = wave * 3 + q;
        accx[q] = wmma_bf16(fy, ldfrag(sWx, kt * 24 + nt), accx[q]);
        acch[q] = wmma_bf16(fh, ldfrag(sWh, kt * 24 + nt), acch[q]);
      }
    }
    {
      int cn = lane & 15, cmb = (lane >> 4) * 8;
#pragma unroll
      for (int q = 0; q < 3; ++q) {
        int n = (wave * 3 + q) * 16 + cn;
#pragma unroll
        for (int r = 0; r < 8; ++r) {
          sGx[(cmb + r) * 384 + n] = accx[q][r];
          sGh[(cmb + r) * 384 + n] = acch[q][r];
        }
      }
    }
    __syncthreads();

    // gates: r,z sigmoid; n-gate needs separate xn / hn
    for (int i = tid; i < 16 * 128; i += 256) {
      int m = i >> 7, j = i & 127;
      float tv = ts[(size_t)(b0 + m) * T_ + rev];
      float xr = sGx[m * 384 + j]       + sb[j]       + tv * swt[j];
      float xz = sGx[m * 384 + j + 128] + sb[j + 128] + tv * swt[j + 128];
      float xn = sGx[m * 384 + j + 256] + sb[j + 256] + tv * swt[j + 256];
      float hr = sGh[m * 384 + j];
      float hz = sGh[m * 384 + j + 128];
      float hn = sGh[m * 384 + j + 256];
      float rg = 1.f / (1.f + expf(-(xr + hr)));
      float zg = 1.f / (1.f + expf(-(xz + hz)));
      float ng = tanhf(xn + rg * hn);
      float h2 = (1.f - zg) * ng + zg * shst[i];
      shst[i] = h2;
      sAh[a_sw(j >> 5, m, j & 31)] = f2bf(h2);
    }
    __syncthreads();
  }

  // y0 = h2l_W @ h_final + h2l_b  (16 rows x 64 out, tiny -> VALU dot)
  for (int i = tid; i < 16 * 64; i += 256) {
    int m = i >> 6, o = i & 63;
    float acc = h2lb[o];
    for (int k = 0; k < 128; ++k) acc += h2lW[o * 128 + k] * shst[m * 128 + k];
    y0out[(b0 + m) * 64 + o] = acc;
  }
}

// ---- kernel 2: truncated Euler solve of the latent ODE -------------------
// 4 workgroups x 16 rows, 5 intervals x 400 fine steps. Per step three
// bf16 WMMA GEMMs through the vf MLP; weights resident in LDS.

__global__ void __launch_bounds__(256) euler_vf_kernel(
    const float* __restrict__ ts, const float* __restrict__ us,
    const float* __restrict__ W0, const float* __restrict__ b0v_,
    const float* __restrict__ W1, const float* __restrict__ b1v_,
    const float* __restrict__ W2, const float* __restrict__ b2v_,
    const float* __restrict__ y0in, float* __restrict__ lat)
{
  extern __shared__ __align__(32) char smem_raw[];
  char* p = smem_raw;
  auto alloc = [&](size_t n) { char* r = p; p += (n + 31) & ~size_t(31); return r; };
  __bf16* sW0 = (__bf16*)alloc(3 * 16 * 512 * 2);  // K=96(pad72) N=256
  __bf16* sW1 = (__bf16*)alloc(8 * 16 * 512 * 2);  // K=256 N=256
  __bf16* sW2 = (__bf16*)alloc(8 * 4 * 512 * 2);   // K=256 N=64
  __bf16* sAz = (__bf16*)alloc(3 * 512 * 2);       // A: [y(64) u(8) pad] K=96
  __bf16* sA1 = (__bf16*)alloc(8 * 512 * 2);
  __bf16* sA2 = (__bf16*)alloc(8 * 512 * 2);
  float*  sb0 = (float*)alloc(256 * 4);
  float*  sb1 = (float*)alloc(256 * 4);
  float*  sb2 = (float*)alloc(64 * 4);
  float*  syst = (float*)alloc(16 * 64 * 4);       // current y
  float*  sysv = (float*)alloc(16 * 64 * 4);       // last saved y (carry)
  float*  sbeg = (float*)alloc(5 * 16 * 4);        // interval start times
  float*  sdt0 = (float*)alloc(16 * 4);

  const int tid = threadIdx.x, wave = tid >> 5, lane = tid & 31;
  const int b0 = blockIdx.x * 16;

  for (int i = tid; i < 256 * 96; i += 256) {
    int n = i / 96, k = i % 96;
    sW0[b_sw((k >> 5) * 16 + (n >> 4), k & 31, n & 15)] =
        f2bf(k < 72 ? W0[n * 72 + k] : 0.f);
  }
  for (int i = tid; i < 256 * 256; i += 256) {
    int n = i >> 8, k = i & 255;
    sW1[b_sw((k >> 5) * 16 + (n >> 4), k & 31, n & 15)] = f2bf(W1[n * 256 + k]);
  }
  for (int i = tid; i < 64 * 256; i += 256) {
    int n = i >> 8, k = i & 255;
    sW2[b_sw((k >> 5) * 4 + (n >> 4), k & 31, n & 15)] = f2bf(W2[n * 256 + k]);
  }
  if (tid < 256) { sb0[tid] = b0v_[tid]; sb1[tid] = b1v_[tid]; }
  if (tid < 64) sb2[tid] = b2v_[tid];
  for (int i = tid; i < 3 * 512; i += 256) sAz[i] = f2bf(0.f);  // zero pad k>=72
  for (int i = tid; i < 16 * 64; i += 256) {
    int m = i >> 6, o = i & 63;
    float v = y0in[(b0 + m) * 64 + o];
    syst[i] = v; sysv[i] = v;
    sAz[a_sw(o >> 5, m, o & 31)] = f2bf(v);
    lat[((size_t)(b0 + m) * T_ + 0) * 64 + o] = v;   // interval-0 save @ t=0
  }
  if (tid < 16) {
    int m = tid;
    float t0 = ts[(size_t)(b0 + m) * T_];
    float t1 = ts[(size_t)(b0 + m) * T_ + 1];
    float d0 = (t1 - t0) * 0.5f;                     // Euler dt = half grid dt
    sdt0[m] = d0;
    sbeg[m] = t0;
    for (int iv = 1; iv < 5; ++iv)
      sbeg[iv * 16 + m] = ts[(size_t)(b0 + m) * T_ + iv * 200 - 1] + d0;
  }
  __syncthreads();

  for (int iv = 0; iv < 5; ++iv) {
    for (int e = 1; e <= 400; ++e) {
      // control interpolation u(t) into Az (k = 64..71)
      if (tid < 128) {
        int m = tid >> 3, c = tid & 7;
        float d0 = sdt0[m], dg = 2.f * d0;
        float t = sbeg[iv * 16 + m] + (float)(e - 1) * d0;
        float t0row = sbeg[m];
        int iu = (int)floorf((t - t0row) / dg);
        iu = iu < 0 ? 0 : (iu > T_ - 2 ? T_ - 2 : iu);
        float w = (t - (t0row + iu * dg)) / dg;
        size_t base = (size_t)(b0 + m) * T_ * C_ + (size_t)iu * C_ + c;
        float u0 = us[base], u1 = us[base + C_];
        sAz[a_sw(2, m, c)] = f2bf(u0 + w * (u1 - u0));
      }
      __syncthreads();

      {  // layer 0: [16,96] x [96,256], tanh
        v8f acc[2]; acc[0] = vzero(); acc[1] = vzero();
#pragma unroll
        for (int kt = 0; kt < 3; ++kt) {
          v16bf a = ldfrag(sAz, kt);
#pragma unroll
          for (int q = 0; q < 2; ++q)
            acc[q] = wmma_bf16(a, ldfrag(sW0, kt * 16 + wave * 2 + q), acc[q]);
        }
        int cn = lane & 15, cmb = (lane >> 4) * 8;
#pragma unroll
        for (int q = 0; q < 2; ++q) {
          int n = (wave * 2 + q) * 16 + cn;
#pragma unroll
          for (int r = 0; r < 8; ++r)
            sA1[a_sw(n >> 5, cmb + r, n & 31)] = f2bf(tanhf(acc[q][r] + sb0[n]));
        }
      }
      __syncthreads();

      {  // layer 1: [16,256] x [256,256], tanh
        v8f acc[2]; acc[0] = vzero(); acc[1] = vzero();
#pragma unroll
        for (int kt = 0; kt < 8; ++kt) {
          v16bf a = ldfrag(sA1, kt);
#pragma unroll
          for (int q = 0; q < 2; ++q)
            acc[q] = wmma_bf16(a, ldfrag(sW1, kt * 16 + wave * 2 + q), acc[q]);
        }
        int cn = lane & 15, cmb = (lane >> 4) * 8;
#pragma unroll
        for (int q = 0; q < 2; ++q) {
          int n = (wave * 2 + q) * 16 + cn;
#pragma unroll
          for (int r = 0; r < 8; ++r)
            sA2[a_sw(n >> 5, cmb + r, n & 31)] = f2bf(tanhf(acc[q][r] + sb1[n]));
        }
      }
      __syncthreads();

      // layer 2 + Euler update + saves: waves 0..3 (wave-uniform branch)
      if (wave < 4) {
        v8f acc = vzero();
#pragma unroll
        for (int kt = 0; kt < 8; ++kt)
          acc = wmma_bf16(ldfrag(sA2, kt), ldfrag(sW2, kt * 4 + wave), acc);
        int cn = lane & 15, cmb = (lane >> 4) * 8;
        int n = wave * 16 + cn;
        bool dosave = (iv == 0) ? (((e & 1) == 0) && e <= 398) : ((e & 1) == 1);
        int tg = (iv == 0) ? (e >> 1) : (iv * 200 + ((e - 1) >> 1));
#pragma unroll
        for (int r = 0; r < 8; ++r) {
          int m = cmb + r;
          float ynew = syst[m * 64 + n] + sdt0[m] * (acc[r] + sb2[n]);
          syst[m * 64 + n] = ynew;
          sAz[a_sw(n >> 5, m, n & 31)] = f2bf(ynew);
          if (dosave) {
            lat[((size_t)(b0 + m) * T_ + tg) * 64 + n] = ynew;
            sysv[m * 64 + n] = ynew;
          }
        }
      }
      __syncthreads();
    }
    if (iv < 4) {  // carry = last *saved* state (stop_gradient path)
      for (int i = tid; i < 16 * 64; i += 256) {
        int m = i >> 6, o = i & 63;
        float v = sysv[i];
        syst[i] = v;
        sAz[a_sw(o >> 5, m, o & 31)] = f2bf(v);
      }
      __syncthreads();
    }
  }
}

// ---- kernel 3: fused readout (hid -> softplus rates + behavior) ----------
// 1000 workgroups x 64 rows of the flattened [B*T, L] latent trajectory.

__global__ void __launch_bounds__(256) readout_kernel(
    const float* __restrict__ lat, const float* __restrict__ l2hW,
    const float* __restrict__ l2hb, const float* __restrict__ rW,
    const float* __restrict__ rb, const float* __restrict__ bW,
    const float* __restrict__ bb, float* __restrict__ out)
{
  extern __shared__ __align__(32) char smem_raw[];
  char* p = smem_raw;
  auto alloc = [&](size_t n) { char* r = p; p += (n + 31) & ~size_t(31); return r; };
  __bf16* sWl = (__bf16*)alloc(2 * 8 * 512 * 2);  // K=64  N=128
  __bf16* sWr = (__bf16*)alloc(4 * 8 * 512 * 2);  // K=128 N=128
  __bf16* sWb = (__bf16*)alloc(4 * 1 * 512 * 2);  // K=128 N=16(pad 4)
  __bf16* sAl = (__bf16*)alloc(4 * 2 * 512 * 2);  // lat  [64,64]
  __bf16* sAh = (__bf16*)alloc(4 * 4 * 512 * 2);  // hid  [64,128]
  float*  sbl = (float*)alloc(128 * 4);
  float*  sbr = (float*)alloc(128 * 4);
  float*  sbb = (float*)alloc(16 * 4);

  const int tid = threadIdx.x, wave = tid >> 5, lane = tid & 31;
  const size_t row0 = (size_t)blockIdx.x * 64;

  for (int i = tid; i < 128 * 64; i += 256) {
    int n = i >> 6, k = i & 63;
    sWl[b_sw((k >> 5) * 8 + (n >> 4), k & 31, n & 15)] = f2bf(l2hW[n * 64 + k]);
  }
  for (int i = tid; i < 128 * 128; i += 256) {
    int n = i >> 7, k = i & 127;
    sWr[b_sw((k >> 5) * 8 + (n >> 4), k & 31, n & 15)] = f2bf(rW[n * 128 + k]);
  }
  for (int i = tid; i < 16 * 128; i += 256) {
    int n = i >> 7, k = i & 127;
    sWb[b_sw((k >> 5), k & 31, n & 15)] = f2bf(n < 4 ? bW[n * 128 + k] : 0.f);
  }
  if (tid < 128) { sbl[tid] = l2hb[tid]; sbr[tid] = rb[tid]; }
  if (tid < 16) sbb[tid] = tid < 4 ? bb[tid] : 0.f;
  for (int i = tid; i < 64 * 64; i += 256) {
    int m = i >> 6, k = i & 63;
    sAl[a_sw((m >> 4) * 2 + (k >> 5), m & 15, k & 31)] = f2bf(lat[(row0 + m) * 64 + k]);
  }
  __syncthreads();

  // hid = lat @ l2h^T + b : 4 m-tiles x 8 n-tiles over 8 waves (4 pairs each)
#pragma unroll
  for (int q = 0; q < 4; ++q) {
    int pi = wave * 4 + q, mt = pi >> 3, nt = pi & 7;
    v8f acc = vzero();
#pragma unroll
    for (int kt = 0; kt < 2; ++kt)
      acc = wmma_bf16(ldfrag(sAl, mt * 2 + kt), ldfrag(sWl, kt * 8 + nt), acc);
    int cn = lane & 15, cmb = (lane >> 4) * 8;
    int n = nt * 16 + cn;
#pragma unroll
    for (int r = 0; r < 8; ++r)
      sAh[a_sw(mt * 4 + (n >> 5), cmb + r, n & 31)] = f2bf(acc[r] + sbl[n]);
  }
  __syncthreads();

  // rates = softplus(hid @ roRate^T + b)
#pragma unroll
  for (int q = 0; q < 4; ++q) {
    int pi = wave * 4 + q, mt = pi >> 3, nt = pi & 7;
    v8f acc = vzero();
#pragma unroll
    for (int kt = 0; kt < 4; ++kt)
      acc = wmma_bf16(ldfrag(sAh, mt * 4 + kt), ldfrag(sWr, kt * 8 + nt), acc);
    int cn = lane & 15, cmb = (lane >> 4) * 8;
    int n = nt * 16 + cn;
#pragma unroll
    for (int r = 0; r < 8; ++r) {
      float v = acc[r] + sbr[n];
      float sp = v > 20.f ? v : log1pf(expf(v));
      size_t row = row0 + mt * 16 + cmb + r;
      out[row * N_ + n] = sp;
    }
  }

  // beh = hid @ roBeh^T + b (N padded 4->16); waves 0..3, wave-uniform
  if (wave < 4) {
    int mt = wave;
    v8f acc = vzero();
#pragma unroll
    for (int kt = 0; kt < 4; ++kt)
      acc = wmma_bf16(ldfrag(sAh, mt * 4 + kt), ldfrag(sWb, kt), acc);
    int cn = lane & 15, cmb = (lane >> 4) * 8;
    if (cn < 4) {
#pragma unroll
      for (int r = 0; r < 8; ++r) {
        size_t row = row0 + mt * 16 + cmb + r;
        out[(size_t)B_ * T_ * N_ + row * BH_ + cn] = acc[r] + sbb[cn];
      }
    }
  }
}

// ---- host launcher -------------------------------------------------------

extern "C" void kernel_launch(void* const* d_in, const int* in_sizes, int n_in,
                              void* d_out, int out_size, void* d_ws, size_t ws_size,
                              hipStream_t stream) {
  const float* ts        = (const float*)d_in[0];
  const float* ys        = (const float*)d_in[1];
  const float* us        = (const float*)d_in[2];
  const float* enc_Wx    = (const float*)d_in[3];
  const float* enc_Wh    = (const float*)d_in[4];
  const float* enc_b     = (const float*)d_in[5];
  const float* h2l_W     = (const float*)d_in[6];
  const float* h2l_b     = (const float*)d_in[7];
  const float* vf_W0     = (const float*)d_in[8];
  const float* vf_b0     = (const float*)d_in[9];
  const float* vf_W1     = (const float*)d_in[10];
  const float* vf_b1     = (const float*)d_in[11];
  const float* vf_W2     = (const float*)d_in[12];
  const float* vf_b2     = (const float*)d_in[13];
  const float* l2h_W     = (const float*)d_in[14];
  const float* l2h_b     = (const float*)d_in[15];
  const float* ro_rate_W = (const float*)d_in[16];
  const float* ro_rate_b = (const float*)d_in[17];
  const float* ro_beh_W  = (const float*)d_in[18];
  const float* ro_beh_b  = (const float*)d_in[19];

  float* lat = (float*)d_ws;                        // [B*T, L] fp32
  float* y0  = lat + (size_t)B_ * T_ * L_;          // [B, L]  fp32

  const size_t smem1 = 2*(4*24*512*2) + 2*(16*384*4) + 2*(4*512*2)
                     + 16*128*4 + 2*(384*4);                         // ~259 KB
  const size_t smem2 = (3*16*512*2) + (8*16*512*2) + (8*4*512*2)
                     + (3*512*2) + 2*(8*512*2)
                     + 256*4 + 256*4 + 64*4 + 2*(16*64*4)
                     + 5*16*4 + 16*4;                                // ~238 KB
  const size_t smem3 = (2*8*512*2) + (4*8*512*2) + (4*1*512*2)
                     + (4*2*512*2) + (4*4*512*2)
                     + 128*4 + 128*4 + 16*4;                         // ~77 KB

  gru_encoder_kernel<<<4, 256, smem1, stream>>>(ts, ys, enc_Wx, enc_Wh, enc_b,
                                                h2l_W, h2l_b, y0);
  euler_vf_kernel<<<4, 256, smem2, stream>>>(ts, us, vf_W0, vf_b0, vf_W1, vf_b1,
                                             vf_W2, vf_b2, y0, lat);
  readout_kernel<<<1000, 256, smem3, stream>>>(lat, l2h_W, l2h_b, ro_rate_W,
                                               ro_rate_b, ro_beh_W, ro_beh_b,
                                               (float*)d_out);
}
